// Model_87282325389459
// MI455X (gfx1250) — compile-verified, matrix operand order checked by reading, not run on hardware
//
#include <hip/hip_runtime.h>
#include <cmath>

typedef __attribute__((ext_vector_type(16))) _Float16     v16h;
typedef __attribute__((ext_vector_type(8)))  float        v8f;
typedef __attribute__((ext_vector_type(4)))  unsigned int v4u;
typedef __attribute__((ext_vector_type(8)))  int          v8i;
typedef __attribute__((ext_vector_type(4)))  int          v4i;

#define S_TOKENS 8192
#define D_PAD    128
#define DK_OUT   100
#define KBLK     32           /* keys per iteration */
#define NITER    (S_TOKENS / KBLK)
#define LOG2E    1.44269504088896340736f
#define INV_SCALE 0.1f        /* 1/sqrt(100) */

// ---- generic 16x32 f16 WMMA A/B fragment loader (two 16B loads per lane) ----
__device__ __forceinline__ v16h load_frag16x32(const _Float16* __restrict__ base,
                                               int row, int k0, int ld) {
  union { v16h v; uint4 u[2]; } r;
  const int lane = threadIdx.x & 31;
  const int kb = (lane >> 4) << 3;
  const _Float16* p = base + (size_t)row * (size_t)ld + (size_t)(k0 + kb);
  r.u[0] = *reinterpret_cast<const uint4*>(p);
  r.u[1] = *reinterpret_cast<const uint4*>(p + 16);
  return r.v;
}

// Same fragment layout but from an LDS-resident tile (lowers to ds_load_b128).
__device__ __forceinline__ v16h lds_frag16x32(const _Float16* p) {
  union { v16h v; uint4 u[2]; } r;
  r.u[0] = *reinterpret_cast<const uint4*>(p);
  r.u[1] = *reinterpret_cast<const uint4*>(p + 16);
  return r.v;
}

// ---- Tensor Data Mover: async 2D f16 tile load Global -> LDS (D# per ISA §8) ----
__device__ __forceinline__ void tdm_load_2d_f16(unsigned int lds_off, const void* gptr,
                                                unsigned int tensor_d0, unsigned int tensor_d1,
                                                unsigned int tile_d0, unsigned int tile_d1,
                                                unsigned long long stride0) {
  const unsigned long long ga = (unsigned long long)(uintptr_t)gptr;
  v4u g0;
  g0[0] = 1u;                                               // count=1, user descriptor
  g0[1] = lds_off;                                          // lds_addr (bytes)
  g0[2] = (unsigned int)ga;                                 // global_addr[31:0]
  g0[3] = (unsigned int)((ga >> 32) & 0x1FFFFFFull)         // global_addr[56:32]
          | (2u << 30);                                     // type = 2 ("image")
  v8i g1;
  g1[0] = (int)(1u << 16);                                  // wg_mask=0, data_size=1 (2B)
  g1[1] = (int)((tensor_d0 & 0xFFFFu) << 16);               // tensor_dim0[15:0]
  g1[2] = (int)((tensor_d0 >> 16) | ((tensor_d1 & 0xFFFFu) << 16));
  g1[3] = (int)((tensor_d1 >> 16) | (tile_d0 << 16));       // tensor_dim1 hi | tile_dim0
  g1[4] = (int)(tile_d1 & 0xFFFFu);                         // tile_dim1, tile_dim2=0
  g1[5] = (int)(unsigned int)(stride0 & 0xFFFFFFFFull);     // dim0_stride[31:0]
  g1[6] = (int)(unsigned int)((stride0 >> 32) & 0xFFFFull); // dim0_stride[47:32]
  g1[7] = 0;                                                // dim1_stride (unused, 2D tile)
  const v4i z4 = {0, 0, 0, 0};
  const v8i z8 = {0, 0, 0, 0, 0, 0, 0, 0};
  __builtin_amdgcn_tensor_load_to_lds(g0, g1, z4, z4, z8, 0);
}

// ---------------- Kernel 1: QKV projection (fp32 -> f16, padded) ----------------
__global__ void qkv_proj_kernel(const float* __restrict__ x1, const float* __restrict__ x2,
                                const float* __restrict__ Wq, const float* __restrict__ bq,
                                const float* __restrict__ Wk, const float* __restrict__ bk,
                                const float* __restrict__ Wv, const float* __restrict__ bv,
                                _Float16* __restrict__ qh, _Float16* __restrict__ kh,
                                _Float16* __restrict__ vT) {
  const int idx = blockIdx.x * blockDim.x + threadIdx.x;
  const int row = idx >> 7;
  const int col = idx & (D_PAD - 1);
  float aq = 0.f, ak = 0.f, av = 0.f;
  if (col < DK_OUT) {
    aq = bq[col]; ak = bk[col]; av = bv[col];
    const float* __restrict__ p1 = x1 + (size_t)row * D_PAD;
    const float* __restrict__ p2 = x2 + (size_t)row * D_PAD;
#pragma unroll 4
    for (int d = 0; d < D_PAD; ++d) {
      const float w1 = p1[d], w2 = p2[d];
      aq = fmaf(w1, Wq[d * DK_OUT + col], aq);
      ak = fmaf(w2, Wk[d * DK_OUT + col], ak);
      av = fmaf(w2, Wv[d * DK_OUT + col], av);
    }
  }
  qh[(size_t)row * D_PAD + col] = (_Float16)aq;
  kh[(size_t)row * D_PAD + col] = (_Float16)ak;
  vT[(size_t)col * S_TOKENS + row] = (_Float16)av;
}

// ---------------- Kernel 2: fused flash attention, WMMA f16 + TDM staging ----------------
__global__ void __launch_bounds__(128, 1)
flash_attn_kernel(const _Float16* __restrict__ qh, const _Float16* __restrict__ kh,
                  const _Float16* __restrict__ vT, float* __restrict__ out) {
  // Double-buffered LDS tiles: K tile 32x128 f16 (8KB), V^T tile 128x32 f16 (8KB).
  __shared__ _Float16 smK[2][KBLK * D_PAD];
  __shared__ _Float16 smV[2][D_PAD * KBLK];

  const int lane = threadIdx.x & 31;
  const int wave = threadIdx.x >> 5;
  const int qt   = blockIdx.x * 4 + wave;
  const int ln   = lane & 15;
  const int kb8  = (lane >> 4) << 3;

  // Provably wave-uniform DMA-issuer guard (SGPR compare -> scalar branch);
  // TDM ops ignore EXEC, so we must branch around them, not mask them.
  const bool dma_wave = ((__builtin_amdgcn_readfirstlane((int)threadIdx.x) >> 5) == 0);

  // Q^T B-fragments (B[d,q] = Q[q,d]); column q = ln. Register-resident.
  v16h qf[4];
#pragma unroll
  for (int c = 0; c < 4; ++c)
    qf[c] = load_frag16x32(qh, qt * 16 + ln, c * 32, D_PAD);

  v8f o[8] = {};
  float m = -INFINITY, l = 0.f;

  // Prologue: DMA tile 0 into buffer 0 via the Tensor Data Mover.
  if (dma_wave) {
    tdm_load_2d_f16((unsigned int)(uintptr_t)&smK[0][0], kh,
                    D_PAD, S_TOKENS, D_PAD, KBLK, D_PAD);
    tdm_load_2d_f16((unsigned int)(uintptr_t)&smV[0][0], vT,
                    S_TOKENS, D_PAD, KBLK, D_PAD, S_TOKENS);
  }

  for (int iter = 0; iter < NITER; ++iter) {
    const int cur = iter & 1;
    const int kb0 = iter * KBLK;
    if (dma_wave) {
      if (iter + 1 < NITER) {
        const int nxt = cur ^ 1;
        tdm_load_2d_f16((unsigned int)(uintptr_t)&smK[nxt][0],
                        kh + (size_t)(kb0 + KBLK) * D_PAD,
                        D_PAD, S_TOKENS, D_PAD, KBLK, D_PAD);
        tdm_load_2d_f16((unsigned int)(uintptr_t)&smV[nxt][0],
                        vT + (size_t)(kb0 + KBLK),
                        S_TOKENS, D_PAD, KBLK, D_PAD, S_TOKENS);
        __builtin_amdgcn_s_wait_tensorcnt((short)2);  // current pair done, next in flight
      } else {
        __builtin_amdgcn_s_wait_tensorcnt((short)0);
      }
    }
    __syncthreads();   // publish current LDS tiles to all waves

    const _Float16* __restrict__ kt = &smK[cur][0];  // [32][128]
    const _Float16* __restrict__ vt = &smV[cur][0];  // [128][32]

    // Preload ALL K fragments first (independent regs) so the ds_load_b128s
    // clause/pipeline instead of a full s_wait_dscnt 0 before every WMMA.
    v16h ak0[4], ak1[4];
#pragma unroll
    for (int c = 0; c < 4; ++c) {
      ak0[c] = lds_frag16x32(kt + (ln)      * D_PAD + c * 32 + kb8);
      ak1[c] = lds_frag16x32(kt + (16 + ln) * D_PAD + c * 32 + kb8);
    }

    v8f s0 = {}; v8f s1 = {};
#pragma unroll
    for (int c = 0; c < 4; ++c) {
      s0 = __builtin_amdgcn_wmma_f32_16x16x32_f16(false, ak0[c], false, qf[c], (short)0, s0, false, false);
      s1 = __builtin_amdgcn_wmma_f32_16x16x32_f16(false, ak1[c], false, qf[c], (short)0, s1, false, false);
    }

    // Issue V^T fragment loads now; their DS latency hides under softmax VALU.
    v16h av[8];
#pragma unroll
    for (int t = 0; t < 8; ++t)
      av[t] = lds_frag16x32(vt + (t * 16 + ln) * KBLK + kb8);

    // ---- online softmax (per query column = lane; halves combined via shfl) ----
    float mx = -INFINITY;
#pragma unroll
    for (int j = 0; j < 8; ++j) { mx = fmaxf(mx, s0[j]); mx = fmaxf(mx, s1[j]); }
    mx *= INV_SCALE;
    mx = fmaxf(mx, __shfl_xor(mx, 16, 32));
    const float mnew  = fmaxf(m, mx);
    const float alpha = exp2f((m - mnew) * LOG2E);

    float p0[8], p1[8], ls = 0.f;
#pragma unroll
    for (int j = 0; j < 8; ++j) {
      p0[j] = exp2f((s0[j] * INV_SCALE - mnew) * LOG2E);
      p1[j] = exp2f((s1[j] * INV_SCALE - mnew) * LOG2E);
      ls += p0[j] + p1[j];
    }
    ls += __shfl_xor(ls, 16, 32);
    l = l * alpha + ls;
    m = mnew;

#pragma unroll
    for (int t = 0; t < 8; ++t)
#pragma unroll
      for (int j = 0; j < 8; ++j) o[t][j] *= alpha;

    // P^T packs register-for-register into the B-operand layout.
    v16h bp;
#pragma unroll
    for (int j = 0; j < 8; ++j) {
      bp[j]     = (_Float16)p0[j];
      bp[8 + j] = (_Float16)p1[j];
    }

    // O^T += V^T_chunk (A, preloaded from LDS) @ P^T (B)
#pragma unroll
    for (int t = 0; t < 8; ++t)
      o[t] = __builtin_amdgcn_wmma_f32_16x16x32_f16(false, av[t], false, bp, (short)0, o[t], false, false);

    __syncthreads();   // all waves done with current tiles before overwrite
  }

  // ---- epilogue: O[q, d] = O^T[d, q] / l, store only d < 100 ----
  const float linv = 1.0f / l;
  const int q = qt * 16 + ln;
#pragma unroll
  for (int t = 0; t < 8; ++t) {
#pragma unroll
    for (int j = 0; j < 8; ++j) {
      const int d = t * 16 + kb8 + j;
      if (d < DK_OUT) out[(size_t)q * DK_OUT + d] = o[t][j] * linv;
    }
  }
}

extern "C" void kernel_launch(void* const* d_in, const int* in_sizes, int n_in,
                              void* d_out, int out_size, void* d_ws, size_t ws_size,
                              hipStream_t stream) {
  const float* x1 = (const float*)d_in[0];
  const float* x2 = (const float*)d_in[1];
  const float* Wq = (const float*)d_in[2];
  const float* bq = (const float*)d_in[3];
  const float* Wk = (const float*)d_in[4];
  const float* bk = (const float*)d_in[5];
  const float* Wv = (const float*)d_in[6];
  const float* bv = (const float*)d_in[7];
  float* out = (float*)d_out;

  _Float16* qh = (_Float16*)d_ws;
  _Float16* kh = qh + (size_t)S_TOKENS * D_PAD;
  _Float16* vT = kh + (size_t)S_TOKENS * D_PAD;

  qkv_proj_kernel<<<(S_TOKENS * D_PAD) / 256, 256, 0, stream>>>(
      x1, x2, Wq, bq, Wk, bk, Wv, bv, qh, kh, vT);

  flash_attn_kernel<<<S_TOKENS / 16 / 4, 128, 0, stream>>>(qh, kh, vT, out);
}
